// GCN_51960514347448
// MI455X (gfx1250) — matrix operand order repaired and checked by reading, not compiled, not run on hardware
//
#include <hip/hip_runtime.h>

#define DIM 128
#define TPB 256

typedef __attribute__((ext_vector_type(16))) __bf16 v16bf;
typedef __attribute__((ext_vector_type(8)))  float  v8f;
typedef __attribute__((ext_vector_type(4)))  int    v4i;

union BFrag { v16bf bf; unsigned int u[8]; uint4 q[2]; };

#if defined(__HIP_DEVICE_COMPILE__) && defined(__gfx1250__) && \
    __has_builtin(__builtin_amdgcn_global_load_async_to_lds_b128) && \
    __has_builtin(__builtin_amdgcn_s_wait_asynccnt)
#define USE_ASYNC_LDS 1
#else
#define USE_ASYNC_LDS 0
#endif

__device__ __forceinline__ unsigned int f2bf(float f) {
    unsigned int u = __float_as_uint(f);
    unsigned int r = (u + 0x7FFFu + ((u >> 16) & 1u)) >> 16;   // RNE f32->bf16 (setup path only)
    return r & 0xFFFFu;
}
__device__ __forceinline__ void atomAddF(float* p, float v) {
    unsafeAtomicAdd(p, v);   // -> global_atomic_add_f32
}

// ---------------- setup kernels ----------------

__global__ __launch_bounds__(TPB) void k_set1(float* deg, int n) {
    int i = blockIdx.x * TPB + threadIdx.x;
    if (i < n) deg[i] = 1.0f;                  // self-loop
}

__global__ __launch_bounds__(TPB) void k_degedge(const int* __restrict__ dst, float* deg, int E) {
    int e = blockIdx.x * TPB + threadIdx.x;
    if (e < E) atomAddF(&deg[dst[e]], 1.0f);
}

__global__ __launch_bounds__(TPB) void k_rsqrt(float* dis, int n) {
    int i = blockIdx.x * TPB + threadIdx.x;
    if (i < n) dis[i] = rsqrtf(dis[i]);        // deg >= 1 always (self loop)
}

__global__ __launch_bounds__(TPB) void k_norm(const int* __restrict__ src, const int* __restrict__ dst,
                                              const float* __restrict__ dis, float* __restrict__ norm, int E) {
    int e = blockIdx.x * TPB + threadIdx.x;
    if (e < E) norm[e] = dis[src[e]] * dis[dst[e]];
}

// Pack W[k,j] (f32 row-major, 16 matrices of 128x128) into bf16 WMMA B-fragment order:
// dest dword index: conv*8192 + ((kt*8+jt)*32 + lane)*8 + d
// lane holds col j = jt*16 + (lane&15), K pair k = kt*32 + (lane>>4)*16 + 2d
__global__ __launch_bounds__(TPB) void k_packw(const float* __restrict__ W1, const float* __restrict__ W2,
                                               unsigned int* __restrict__ wpk) {
    int idx = blockIdx.x * TPB + threadIdx.x;   // 16*8192 = 131072 threads
    int d    =  idx        & 7;
    int lane = (idx >> 3)  & 31;
    int jt   = (idx >> 8)  & 7;
    int kt   = (idx >> 11) & 3;
    int conv =  idx >> 13;
    int layer = conv >> 1, which = conv & 1;
    const float* Ws = (which ? W2 : W1) + (size_t)layer * DIM * DIM;
    int k = kt * 32 + (lane >> 4) * 16 + 2 * d;
    int j = jt * 16 + (lane & 15);
    unsigned int lo = f2bf(Ws[(size_t)k * DIM + j]);
    unsigned int hi = f2bf(Ws[(size_t)(k + 1) * DIM + j]);
    wpk[(size_t)conv * 8192 + (size_t)((kt * 8 + jt) * 32 + lane) * 8 + d] = lo | (hi << 16);
}

// ---------------- GEMM: tmp = act(hin) @ W ; out = bias + dis^2 * tmp ----------------
// One wave computes a 16-row x 128-col strip. N multiple of 16 (50000 = 3125*16).
// W (32KB, packed bf16 fragments) is staged into LDS once per block via the CDNA5
// async global->LDS path, then read back as ds_load_b128 pairs.

__global__ __launch_bounds__(TPB) void k_matmul(const float* __restrict__ hin, int relu,
                                                const unsigned int* __restrict__ wpk,
                                                const float* __restrict__ bias,
                                                const float* __restrict__ dis,
                                                float* __restrict__ tmp, float* __restrict__ outv,
                                                int ntiles) {
    __shared__ v4i sW[2048];                   // 32 KB: one conv's packed weight matrix

    int tid = threadIdx.x;
    // ---- stage W into LDS (all threads participate, then one barrier) ----
#if USE_ASYNC_LDS
    #pragma unroll
    for (int i = 0; i < 8; ++i) {
        int idx = tid + i * TPB;               // v4i index 0..2047
        __builtin_amdgcn_global_load_async_to_lds_b128(
            (__attribute__((address_space(1))) v4i*)((v4i*)wpk + idx),
            (__attribute__((address_space(3))) v4i*)(&sW[idx]),
            0, 0);
    }
    __builtin_amdgcn_s_wait_asynccnt(0);
#else
    #pragma unroll
    for (int i = 0; i < 8; ++i) {
        int idx = tid + i * TPB;
        sW[idx] = ((const v4i*)wpk)[idx];
    }
#endif
    __syncthreads();

    int wave = (blockIdx.x * TPB + tid) >> 5;
    if (wave >= ntiles) return;                // after the only barrier: safe
    int lane  = tid & 31;
    int row0  = wave * 16;
    int arow  = lane & 15;        // A-matrix row within tile
    int khalf = lane >> 4;        // lane half selects K sub-block

    v8f acc[8] = {};
    const float* rowp = hin + (size_t)(row0 + arow) * DIM;

    #pragma unroll
    for (int kt = 0; kt < 4; ++kt) {
        int kbase = kt * 32 + khalf * 16;
        float4 a0 = *(const float4*)(rowp + kbase);
        float4 a1 = *(const float4*)(rowp + kbase + 4);
        float4 a2 = *(const float4*)(rowp + kbase + 8);
        float4 a3 = *(const float4*)(rowp + kbase + 12);
        float av[16] = { a0.x, a0.y, a0.z, a0.w, a1.x, a1.y, a1.z, a1.w,
                         a2.x, a2.y, a2.z, a2.w, a3.x, a3.y, a3.z, a3.w };
        if (relu) {
            #pragma unroll
            for (int i = 0; i < 16; ++i) av[i] = fmaxf(av[i], 0.0f);
        }
        v16bf abf;
        #pragma unroll
        for (int i = 0; i < 16; ++i) abf[i] = (__bf16)av[i];   // hw cvt (v_cvt_pk_bf16_f32)

        #pragma unroll
        for (int jt = 0; jt < 8; ++jt) {
            const uint4* ls = (const uint4*)sW + (size_t)((kt * 8 + jt) * 32 + lane) * 2;
            BFrag bf;
            bf.q[0] = ls[0];                   // ds_load_b128
            bf.q[1] = ls[1];
            acc[jt] = __builtin_amdgcn_wmma_f32_16x16x32_bf16(
                false, abf, false, bf.bf, (short)0, acc[jt], false, false);
        }
    }

    // epilogue: C/D layout -> row = row0 + khalf*8 + r, col = jt*16 + (lane&15)
    int col   = lane & 15;
    int rbase = row0 + khalf * 8;
    float d2[8];
    #pragma unroll
    for (int r = 0; r < 8; ++r) { float dv = dis[rbase + r]; d2[r] = dv * dv; }
    #pragma unroll
    for (int jt = 0; jt < 8; ++jt) {
        float bj = bias[jt * 16 + col];
        #pragma unroll
        for (int r = 0; r < 8; ++r) {
            size_t idx = (size_t)(rbase + r) * DIM + jt * 16 + col;
            float v = acc[jt][r];
            tmp[idx]  = v;
            outv[idx] = bj + d2[r] * v;        // bias + self-loop contribution
        }
    }
}

// ---------------- edge scatter: out[dst] += norm_e * tmp[src] ----------------
// One edge per wave; each lane moves a float4 (coalesced 512B row) -> 4 hw f32 atomics.

__global__ __launch_bounds__(TPB) void k_scatter(const int* __restrict__ src, const int* __restrict__ dst,
                                                 const float* __restrict__ norm,
                                                 const float* __restrict__ tmp, float* __restrict__ outv, int E) {
    int t = blockIdx.x * TPB + threadIdx.x;
    int e = t >> 5;
    if (e >= E) return;
    int l = t & 31;
    int s = src[e], d = dst[e];
    float nm = norm[e];
    float4 v = ((const float4*)(tmp + (size_t)s * DIM))[l];
    float* op = outv + (size_t)d * DIM + l * 4;
    atomAddF(op + 0, v.x * nm);
    atomAddF(op + 1, v.y * nm);
    atomAddF(op + 2, v.z * nm);
    atomAddF(op + 3, v.w * nm);
}

// ---------------- pooling ----------------

__global__ __launch_bounds__(TPB) void k_zero(float* p, int n) {
    int i = blockIdx.x * TPB + threadIdx.x;
    if (i < n) p[i] = 0.0f;
}

__global__ __launch_bounds__(TPB) void k_pool(const float* __restrict__ h, const int* __restrict__ batch,
                                              float* __restrict__ pool, float* __restrict__ cnt, int N) {
    int t = blockIdx.x * TPB + threadIdx.x;
    int n = t >> 5;
    if (n >= N) return;
    int l = t & 31;
    int g = batch[n];
    float4 v = ((const float4*)(h + (size_t)n * DIM))[l];
    float* pp = pool + (size_t)g * DIM + l * 4;
    atomAddF(pp + 0, v.x);
    atomAddF(pp + 1, v.y);
    atomAddF(pp + 2, v.z);
    atomAddF(pp + 3, v.w);
    if (l == 0) atomAddF(&cnt[g], 1.0f);
}

__global__ __launch_bounds__(TPB) void k_final(const float* __restrict__ pool, const float* __restrict__ cnt,
                                               float* __restrict__ out, int n) {
    int i = blockIdx.x * TPB + threadIdx.x;
    if (i < n) out[i] = pool[i] / fmaxf(cnt[i >> 7], 1.0f);
}

// ---------------- host ----------------

extern "C" void kernel_launch(void* const* d_in, const int* in_sizes, int n_in,
                              void* d_out, int out_size, void* d_ws, size_t ws_size,
                              hipStream_t stream) {
    const float* x     = (const float*)d_in[0];
    const int*   ei    = (const int*)  d_in[1];
    const int*   batch = (const int*)  d_in[2];
    const float* W1    = (const float*)d_in[3];
    const float* b1    = (const float*)d_in[4];
    const float* W2    = (const float*)d_in[5];
    const float* b2    = (const float*)d_in[6];

    const int N = in_sizes[0] / DIM;      // 50000 (multiple of 16)
    const int E = in_sizes[1] / 2;        // 600000
    const int G = 64;
    const int* src = ei;
    const int* dst = ei + E;

    char* ws = (char*)d_ws;
    size_t off = 0;
    auto alloc = [&](size_t bytes) -> char* {
        char* p = ws + off;
        off += (bytes + 255) & ~(size_t)255;
        return p;
    };
    float*        dis  = (float*)alloc((size_t)N * 4);            // deg then dis, in place
    float*        norm = (float*)alloc((size_t)E * 4);
    float*        bufA = (float*)alloc((size_t)N * DIM * 4);
    float*        bufB = (float*)alloc((size_t)N * DIM * 4);      // tmp (= h@W)
    float*        bufC = (float*)alloc((size_t)N * DIM * 4);
    unsigned int* wpk  = (unsigned int*)alloc((size_t)16 * 8192 * 4);
    float*        pool = (float*)alloc((size_t)(G * DIM + G) * 4);
    float*        cnt  = pool + G * DIM;

    int gN = (N + TPB - 1) / TPB;
    int gE = (E + TPB - 1) / TPB;

    // degrees -> dis -> edge norms (once, reused by all 16 convs)
    k_set1   <<<gN, TPB, 0, stream>>>(dis, N);
    k_degedge<<<gE, TPB, 0, stream>>>(dst, dis, E);
    k_rsqrt  <<<gN, TPB, 0, stream>>>(dis, N);
    k_norm   <<<gE, TPB, 0, stream>>>(src, dst, dis, norm, E);

    // pre-pack all 16 weight matrices into bf16 WMMA fragments
    k_packw<<<(16 * 8192) / TPB, TPB, 0, stream>>>(W1, W2, wpk);

    const int ntiles = N / 16;
    const int gMM = (ntiles * 32 + TPB - 1) / TPB;
    const int gSC = (int)(((size_t)E * 32 + TPB - 1) / TPB);

    const float* hin = x;
    float* outbuf = bufA;
    for (int c = 0; c < 16; ++c) {
        int layer = c >> 1, which = c & 1;
        const unsigned int* wm = wpk + (size_t)c * 8192;
        const float* bias = (which ? b2 : b1) + (size_t)layer * DIM;
        // ReLU is fused onto the *load* of the second conv's matmul input
        k_matmul <<<gMM, TPB, 0, stream>>>(hin, which, wm, bias, dis, bufB, outbuf, ntiles);
        k_scatter<<<gSC, TPB, 0, stream>>>(src, dst, norm, bufB, outbuf, E);
        hin = outbuf;
        outbuf = (outbuf == bufA) ? bufC : bufA;
    }

    // mean pooling per graph
    k_zero <<<(G * DIM + G + TPB - 1) / TPB, TPB, 0, stream>>>(pool, G * DIM + G);
    k_pool <<<(N * 32 + TPB - 1) / TPB, TPB, 0, stream>>>(hin, batch, pool, cnt, N);
    k_final<<<(G * DIM + TPB - 1) / TPB, TPB, 0, stream>>>(pool, cnt, (float*)d_out, G * DIM);
}
